// ReconstructionModel_29807073034355
// MI455X (gfx1250) — matrix operand order repaired and checked by reading, not compiled, not run on hardware
//
#include <hip/hip_runtime.h>
#include <stdint.h>

#define B_    16
#define L_    4096
#define C_    161
#define SIDE_ 48
#define M_    64
#define NSTEP_ 32
#define WCS   (C_ * SIDE_)   // 7728 elements per [C][S] window

typedef __bf16 bf16_t;
typedef bf16_t v16bf __attribute__((ext_vector_type(16)));
typedef float  v8f   __attribute__((ext_vector_type(8)));

// Param table: jax pytree leaves of params dict (keys sorted):
//  ae1: 6 x { b0..b3, tw, w0..w3 }                -> idx 9*ae + {0..3,4,5..8}
//  down: left/right x { b0..b4, tw, w0..w4 }      -> base 54 + 11*side
//  final: { b1, b2, tw, w1, w2 }                  -> 76..80
//  lin: left/right x { b1, b2, tw, w1, w2 }       -> base 81 + 5*side
struct PT { const float* p[91]; };

__device__ inline unsigned short f2b(float f) {
  return (unsigned short)(__builtin_bit_cast(unsigned int, f) >> 16);
}
__device__ inline float b2f(unsigned short u) {
  return __builtin_bit_cast(float, ((unsigned int)u) << 16);
}
__device__ inline float taylor5(float x, const float* w) {
  return (((w[4] * x + w[3]) * x + w[2]) * x + w[1]) * x + w[0];
}

// ---------------------------------------------------------------- mask prep
__global__ void prep_mask_kernel(const int* __restrict__ mask, int* __restrict__ fl) {
  int b = threadIdx.x;
  if (b >= B_) return;
  const int* m = mask + (size_t)b * L_;
  int first = 0, last = L_ - 1, has = 0;
  for (int i = 0; i < L_; ++i) { if (m[i] != 0) { first = i; has = 1; break; } }
  for (int i = L_ - 1; i >= 0; --i) { if (m[i] != 0) { last = i; break; } }
  fl[b] = first; fl[16 + b] = last; fl[32 + b] = has;
}

__global__ void init_out_kernel(float* __restrict__ out, int n) {
  int i = blockIdx.x * 256 + threadIdx.x;
  if (i < n) out[i] = -1.0f;
}

// ------------------------------------------------------- initial li/ri gather
__global__ void __launch_bounds__(256)
init_li_ri_kernel(const float* __restrict__ x, const int* __restrict__ fl,
                  float* __restrict__ li, float* __restrict__ ri) {
  const int which = blockIdx.x, b = blockIdx.y;
  const int tid = threadIdx.x;
  int first = fl[b], last = fl[16 + b];
  int start, endx; float* dst;
  if (which == 0) { start = max(first - SIDE_, 0); endx = first; dst = li + (size_t)b * WCS; }
  else            { start = last + 1; endx = min(last + 1 + SIDE_, L_); dst = ri + (size_t)b * WCS; }
  for (int i = tid; i < WCS; i += 256) {
    int c = i / SIDE_, s = i - c * SIDE_;
    int idx = start + s;
    float v = 0.f;
    if (idx >= 0 && idx < L_ && idx < endx) v = x[((size_t)b * L_ + idx) * C_ + c];
    dst[i] = v;
  }
}

// --------------------------------------------------------- per-step windows
// w=0: li copy, w=1: lrem gather, w=2: ri copy, w=3: rrem gather
__global__ void __launch_bounds__(256)
stage_windows_kernel(const float* __restrict__ x, const int* __restrict__ fl,
                     const float* __restrict__ li, const float* __restrict__ ri,
                     float* __restrict__ win, int l) {
  const int w = blockIdx.x, b = blockIdx.y;
  const int tid = threadIdx.x;
  float* dst = win + (size_t)(w * B_ + b) * WCS;
  if (w == 0) { const float* s = li + (size_t)b * WCS; for (int i = tid; i < WCS; i += 256) dst[i] = s[i]; return; }
  if (w == 2) { const float* s = ri + (size_t)b * WCS; for (int i = tid; i < WCS; i += 256) dst[i] = s[i]; return; }
  int first = fl[b], last = fl[16 + b];
  int start, endx;
  if (w == 1) { start = last + 2 + l;            endx = min(start + SIDE_, L_ - 1); }
  else        { start = max(first - (2 + l), 0); endx = min(start + SIDE_, L_); }
  for (int i = tid; i < WCS; i += 256) {
    int c = i / SIDE_, s = i - c * SIDE_;
    int idx = start + s;
    float v = 0.f;
    if (idx >= 0 && idx < L_ && idx < endx) v = x[((size_t)b * L_ + idx) * C_ + c];
    dst[i] = v;
  }
}

// ------------------------------------------------- WMMA conv autoencoder stack
// Templated on AE index so the conv kernel width K is a compile-time constant,
// the 4-layer loop is fully unrolled (Cin/Cout/NK/T constants), and each wave's
// accumulator indices are constant -> accumulators stay in VGPRs (no scratch).
// One block per (window w, batch b). 8 waves; a wave owns up to 5 of the
// (CoutTiles x 3) 16x16 output tiles; K-dim streamed in chunks of 32 through
// v_wmma_f32_16x16x32_bf16 with LDS-resident bf16 activations.
template <int AE>
__global__ void __launch_bounds__(256)
ae_stack_kernel(PT pt, const float* __restrict__ win, float* __restrict__ aeout) {
  constexpr int KSc[6] = {25, 21, 17, 13, 9, 5};
  constexpr int K = KSc[AE];
  constexpr int pad = K >> 1;
  constexpr int CINc[4]  = {161, 161, 80, 40};
  constexpr int COUTc[4] = {161, 80, 40, 20};

  const int w = blockIdx.x, b = blockIdx.y;
  const int tid = threadIdx.x;
  const int lane = tid & 31, wv = tid >> 5;

  __shared__ unsigned short actA[WCS];        // bf16 activations (ping)
  __shared__ unsigned short actB[WCS];        // bf16 activations (pong)
  __shared__ unsigned short a_tile[176 * 32]; // weight tile   [Mpad][32]
  __shared__ unsigned short b_tileT[48 * 32]; // im2col tile   [N=48][K=32]

  // stage layer-0 input (f32 global -> bf16 LDS)
  const float* w0 = win + (size_t)(w * B_ + b) * WCS;
  for (int i = tid; i < WCS; i += 256) actA[i] = f2b(w0[i]);
  __syncthreads();

  unsigned short* ain = actA;
  unsigned short* aon = actB;

#pragma unroll
  for (int j = 0; j < 4; ++j) {
    const int Cin = CINc[j], Cout = COUTc[j];
    const int KT = Cin * K;                 // flat reduction length
    const int NK = (KT + 31) >> 5;          // K-chunks of 32
    const int CoutT = (Cout + 15) >> 4;     // M tiles
    const int T = CoutT * 3;                // total 16x16 output tiles
    const int Mpad = CoutT << 4;
    const float* wj  = pt.p[9 * AE + 5 + j];
    const float* bj  = pt.p[9 * AE + j];
    const float* twj = pt.p[9 * AE + 4] + j * 5;

    const v8f vzero = {0.f, 0.f, 0.f, 0.f, 0.f, 0.f, 0.f, 0.f};
    v8f acc[5];
#pragma unroll
    for (int i = 0; i < 5; ++i) acc[i] = vzero;

    for (int kc = 0; kc < NK; ++kc) {
      // --- stage A tile: weights [Mpad][32] as bf16 ---
      for (int idx = tid; idx < Mpad * 32; idx += 256) {
        int m = idx >> 5, kk = idx & 31;
        int flat = (kc << 5) + kk;
        unsigned short v = 0;
        if (m < Cout && flat < KT) {
          int ci = flat / K;            // constant divisor
          int kp = flat - ci * K;
          v = f2b(wj[(size_t)m * KT + (size_t)ci * K + kp]);
        }
        a_tile[idx] = v;
      }
      // --- stage B tile transposed: [n=0..47][k=0..31] bf16, zero-padded ---
      for (int idx = tid; idx < 48 * 32; idx += 256) {
        int s = idx >> 5, kk = idx & 31;
        int flat = (kc << 5) + kk;
        unsigned short v = 0;
        if (flat < KT) {
          int ci = flat / K;
          int kp = flat - ci * K;
          int xs = s + kp - pad;
          if (xs >= 0 && xs < SIDE_) v = ain[ci * SIDE_ + xs];
        }
        b_tileT[idx] = v;
      }
      // prefetch next chunk's weight rows into cache (global_prefetch_b8)
      if (kc + 1 < NK && tid < Mpad) {
        __builtin_prefetch(&wj[(size_t)tid * KT + ((kc + 1) << 5)], 0, 1);
      }
      __syncthreads();

      // --- each wave accumulates its owned tiles (indices compile-time) ---
      const int khiA = (lane >> 4) << 3;   // A frag: lanes 0-15 K{0..7,16..23}
      const int kbB  = (lane >> 4) << 4;   // B frag: same K order as A frag
#pragma unroll
      for (int i = 0; i < 5; ++i) {
        const int t = wv + (i << 3);
        if (t < T) {                       // wave-uniform guard
          int mt = t / 3, nt = t - mt * 3;
          v16bf af, bfr;
          const int ma = ((mt << 4) + (lane & 15)) * 32;
          const int nb = ((nt << 4) + (lane & 15)) * 32;
#pragma unroll
          for (int e = 0; e < 16; ++e) {
            int k = (e & 7) + ((e >> 3) << 4) + khiA;
            af[e]  = __builtin_bit_cast(bf16_t, a_tile[ma + k]);
            bfr[e] = __builtin_bit_cast(bf16_t, b_tileT[nb + k]);
          }
          acc[i] = __builtin_amdgcn_wmma_f32_16x16x32_bf16(
              false, af, false, bfr, (short)0, acc[i], false, false);
        }
      }
      __syncthreads();
    }

    // --- epilogue: bias + taylor activation, write bf16 to pong buffer ---
#pragma unroll
    for (int i = 0; i < 5; ++i) {
      const int t = wv + (i << 3);
      if (t < T) {
        int mt = t / 3, nt = t - mt * 3;
        int s = (nt << 4) + (lane & 15);
        int mofs = (lane >> 4) << 3;       // C/D layout: lanes 16-31 hold M+8
#pragma unroll
        for (int r = 0; r < 8; ++r) {
          int co = (mt << 4) + r + mofs;
          if (co < Cout) {
            float y = taylor5(acc[i][r] + bj[co], twj);
            aon[co * SIDE_ + s] = f2b(y);
          }
        }
      }
    }
    __syncthreads();
    unsigned short* tmp = ain; ain = aon; aon = tmp;
  }

  // ain holds final [20][48]; store per-AE (summed later by down_lin)
  float* dst = aeout + (((size_t)(w * B_ + b) * 6) + AE) * (20 * 48);
  for (int i = tid; i < 20 * 48; i += 256) dst[i] = b2f(ain[i]);
}

// ----------------------------------- dilated convs + mean + 2 linears (small)
__global__ void __launch_bounds__(256)
down_lin_kernel(PT pt, const float* __restrict__ aeout, float* __restrict__ fs) {
  const int w = blockIdx.x, b = blockIdx.y;
  const int tid = threadIdx.x;
  const int side = (w == 0 || w == 3) ? 0 : 1;  // li:left lrem:right ri:right rrem:left

  __shared__ float h0[20 * 48];
  __shared__ float h1[20 * 48];
  __shared__ float mv[20];
  __shared__ float o1[90];

  const float* src = aeout + ((size_t)(w * B_ + b) * 6) * 960;
  for (int i = tid; i < 960; i += 256) {
    float a = 0.f;
    for (int ae = 0; ae < 6; ++ae) a += src[ae * 960 + i];
    h0[i] = a;
  }
  __syncthreads();

  const int dbase = 54 + 11 * side;
  const float* twd = pt.p[dbase + 5];
  const int dil[5] = {1, 2, 4, 8, 16};
  int Sin = 48;
  float* hin = h0; float* hout = h1;
  for (int j = 0; j < 5; ++j) {
    const float* wd = pt.p[dbase + 6 + j];
    const float* bd = pt.p[dbase + j];
    int d = dil[j];
    int Sout = Sin - d;
    for (int idx = tid; idx < 20 * Sout; idx += 256) {
      int co = idx / Sout, s = idx - co * Sout;
      float a = 0.f;
      for (int ci = 0; ci < 20; ++ci) {
        a += wd[(co * 20 + ci) * 2 + 0] * hin[ci * 48 + s];
        a += wd[(co * 20 + ci) * 2 + 1] * hin[ci * 48 + s + d];
      }
      hout[co * 48 + s] = taylor5(a + bd[co], twd + j * 5);
    }
    __syncthreads();
    float* t = hin; hin = hout; hout = t;
    Sin = Sout;
  }
  // Sin == 17; mean over time
  for (int c = tid; c < 20; c += 256) {
    float a = 0.f;
    for (int s = 0; s < 17; ++s) a += hin[c * 48 + s];
    mv[c] = a * (1.0f / 17.0f);
  }
  __syncthreads();

  const int lbase = 81 + 5 * side;
  const float* lb1 = pt.p[lbase + 0];
  const float* lb2 = pt.p[lbase + 1];
  const float* twl = pt.p[lbase + 2];
  const float* lw1 = pt.p[lbase + 3];
  const float* lw2 = pt.p[lbase + 4];
  for (int jj = tid; jj < 90; jj += 256) {
    float a = lb1[jj];
    for (int c = 0; c < 20; ++c) a += mv[c] * lw1[c * 90 + jj];
    o1[jj] = taylor5(a, twl + 0);
  }
  __syncthreads();
  float* dst = fs + (size_t)(w * B_ + b) * C_;
  for (int co = tid; co < C_; co += 256) {
    float a = lb2[co];
    for (int jj = 0; jj < 90; ++jj) a += o1[jj] * lw2[jj * C_ + co];
    dst[co] = taylor5(a, twl + 5);
  }
}

// -------------------- final layer x2, conditional scatter, li/ri shift-update
__global__ void __launch_bounds__(256)
final_update_kernel(PT pt, const float* __restrict__ fs, const int* __restrict__ fl,
                    float* __restrict__ li, float* __restrict__ ri,
                    float* __restrict__ out, int l) {
  const int b = blockIdx.x;
  const int tid = threadIdx.x;
  __shared__ float h1l[241], h1r[241];
  __shared__ float lfv[C_], rfv[C_];
  __shared__ float sb[WCS];

  const float* fb1 = pt.p[76];
  const float* fb2 = pt.p[77];
  const float* ftw = pt.p[78];
  const float* fw1 = pt.p[79];
  const float* fw2 = pt.p[80];
  const float* f0 = fs + (size_t)(0 * B_ + b) * C_;
  const float* f1 = fs + (size_t)(1 * B_ + b) * C_;
  const float* f2v = fs + (size_t)(2 * B_ + b) * C_;
  const float* f3 = fs + (size_t)(3 * B_ + b) * C_;

  for (int j = tid; j < 241; j += 256) {
    float al = fb1[j], ar = fb1[j];
    for (int i = 0; i < C_; ++i) {
      float wlo = fw1[i * 241 + j];
      float whi = fw1[(C_ + i) * 241 + j];
      al += f0[i] * wlo + f1[i] * whi;
      ar += f2v[i] * wlo + f3[i] * whi;
    }
    h1l[j] = taylor5(al, ftw + 0);
    h1r[j] = taylor5(ar, ftw + 0);
  }
  __syncthreads();
  for (int c = tid; c < C_; c += 256) {
    float al = fb2[c], ar = fb2[c];
    for (int j = 0; j < 241; ++j) {
      float wv2 = fw2[j * C_ + c];
      al += h1l[j] * wv2;
      ar += h1r[j] * wv2;
    }
    lfv[c] = fminf(fmaxf(taylor5(al, ftw + 5), 0.f), 6.f);
    rfv[c] = fminf(fmaxf(taylor5(ar, ftw + 5), 0.f), 6.f);
  }
  __syncthreads();

  int first = fl[b], last = fl[16 + b], has = fl[32 + b];
  int ridx = min(last - first - l, M_ - 1 - l);
  bool cond = (has != 0) && (l <= ridx);     // block-uniform
  int ridx_c = min(max(ridx, 0), M_ - 1);

  if (cond) {
    for (int c = tid; c < C_; c += 256) out[((size_t)b * M_ + l) * C_ + c] = lfv[c];
    for (int c = tid; c < C_; c += 256) out[((size_t)b * M_ + ridx_c) * C_ + c] = rfv[c];

    // li = concat(li[1:], lf)  (stage via LDS to avoid RAW across threads)
    float* lb = li + (size_t)b * WCS;
    for (int i = tid; i < WCS; i += 256) sb[i] = lb[i];
    __syncthreads();
    for (int i = tid; i < WCS; i += 256) {
      int c = i / SIDE_, s = i - c * SIDE_;
      lb[i] = (s < SIDE_ - 1) ? sb[c * SIDE_ + s + 1] : lfv[c];
    }
    __syncthreads();
    // ri = concat(rf, ri[:-1])
    float* rb = ri + (size_t)b * WCS;
    for (int i = tid; i < WCS; i += 256) sb[i] = rb[i];
    __syncthreads();
    for (int i = tid; i < WCS; i += 256) {
      int c = i / SIDE_, s = i - c * SIDE_;
      rb[i] = (s == 0) ? rfv[c] : sb[c * SIDE_ + s - 1];
    }
  }
}

// ---------------------------------------------------------------------- host
extern "C" void kernel_launch(void* const* d_in, const int* in_sizes, int n_in,
                              void* d_out, int out_size, void* d_ws, size_t ws_size,
                              hipStream_t stream) {
  (void)in_sizes; (void)ws_size;
  const float* x = (const float*)d_in[0];
  const int* mask = (const int*)d_in[1];

  PT pt;
  int navail = n_in - 2;
  for (int i = 0; i < 91; ++i) pt.p[i] = (const float*)d_in[2 + (i < navail ? i : (navail > 0 ? navail - 1 : 0))];

  int*   fl    = (int*)d_ws;
  float* base  = (float*)d_ws + 64;
  float* li    = base;
  float* ri    = li + (size_t)B_ * WCS;
  float* win   = ri + (size_t)B_ * WCS;
  float* aeout = win + (size_t)4 * B_ * WCS;
  float* fs    = aeout + (size_t)4 * B_ * 6 * 960;

  prep_mask_kernel<<<1, 64, 0, stream>>>(mask, fl);
  init_out_kernel<<<(out_size + 255) / 256, 256, 0, stream>>>((float*)d_out, out_size);
  init_li_ri_kernel<<<dim3(2, B_), 256, 0, stream>>>(x, fl, li, ri);

  for (int l = 0; l < NSTEP_; ++l) {
    stage_windows_kernel<<<dim3(4, B_), 256, 0, stream>>>(x, fl, li, ri, win, l);
    ae_stack_kernel<0><<<dim3(4, B_), 256, 0, stream>>>(pt, win, aeout);
    ae_stack_kernel<1><<<dim3(4, B_), 256, 0, stream>>>(pt, win, aeout);
    ae_stack_kernel<2><<<dim3(4, B_), 256, 0, stream>>>(pt, win, aeout);
    ae_stack_kernel<3><<<dim3(4, B_), 256, 0, stream>>>(pt, win, aeout);
    ae_stack_kernel<4><<<dim3(4, B_), 256, 0, stream>>>(pt, win, aeout);
    ae_stack_kernel<5><<<dim3(4, B_), 256, 0, stream>>>(pt, win, aeout);
    down_lin_kernel<<<dim3(4, B_), 256, 0, stream>>>(pt, aeout, fs);
    final_update_kernel<<<B_, 256, 0, stream>>>(pt, fs, fl, li, ri, (float*)d_out, l);
  }
}
